// AssociativeMemory_13194139533359
// MI455X (gfx1250) — matrix operand order repaired and checked by reading, not compiled, not run on hardware
//
#include <hip/hip_runtime.h>
#include <hip/hip_bf16.h>
#include <math.h>

typedef __attribute__((ext_vector_type(16))) _Float16 v16h;
typedef __attribute__((ext_vector_type(8)))  _Float16 h8;
typedef __attribute__((ext_vector_type(8)))  float    v8f;
typedef __attribute__((ext_vector_type(4)))  float    f4;
typedef __attribute__((ext_vector_type(4)))  unsigned v4u;
typedef __attribute__((ext_vector_type(4)))  int      v4i;
typedef __attribute__((ext_vector_type(8)))  int      v8i;

#define B_    1024
#define D_    256
#define V_    256
#define S_    65536
#define NSC   16            // S chunks (split-softmax)
#define SC    (S_/NSC)      // 4096 slots per chunk
#define BT    64            // query rows per workgroup
#define NBT   (B_/BT)       // 16
#define STRIP 64            // slots per inner strip
#define NSTRIP (SC/STRIP)   // 64
#define NREP  1024

// LDS layout (bytes)
#define QS_STRIDE 264       // halves per q/k row (528B, 16B-mult, 4-bank step)
#define KS_STRIDE 264
#define VT_STRIDE 88        // halves per vT row (176B, 16B-mult, conflict-free)
#define PS_STRIDE 40        // halves per P row  (80B)
#define OFF_QS   0
#define OFF_K0   33792      // K double buffer 0
#define OFF_K1   67584      // K double buffer 1
#define OFF_VT   101376
#define OFF_PS   146432
#define SMEM_BYTES 156672
#define OFF_MRGO 33792      // overlay (post-loop) on K buffers
#define OFF_MRGML 99328

#if defined(__has_builtin)
#if __has_builtin(__builtin_amdgcn_tensor_load_to_lds) && __has_builtin(__builtin_amdgcn_s_wait_tensorcnt)
#define HAVE_TDM 1
#endif
#endif

__device__ __forceinline__ v8f wmma16(v16h a, v16h b, v8f c) {
  return __builtin_amdgcn_wmma_f32_16x16x32_f16(false, a, false, b, (short)0, c,
                                                false, false);
}

#ifdef HAVE_TDM
// TDM: DMA a (rows x dim0) 2-byte-element tile (global row stride = `stride`
// elements) into LDS, inserting `pamt+1` DWORDs of pad after every
// 2^(pint+1) DWORDs so the LDS row stride gets the bank-conflict padding.
__device__ __forceinline__ void tdm_load_2d(const void* gsrc, unsigned lds_off,
                                            unsigned dim0, unsigned rows,
                                            unsigned stride, unsigned pint,
                                            unsigned pamt) {
  unsigned long long ga = (unsigned long long)(uintptr_t)gsrc;
  v4u g0 = { 1u,                                   // count=1 (valid descriptor)
             lds_off,                              // lds_addr
             (unsigned)(ga & 0xFFFFFFFFu),         // global_addr[31:0]
             (unsigned)((ga >> 32) & 0x01FFFFFFu)  // global_addr[56:32]
                 | (2u << 30) };                   // type=2 ("image")
  v8i g1 = { (int)((1u << 16)                      // data_size = 2 bytes
               | (1u << 20)                        // pad_enable
               | (pint << 22)                      // pad_interval
               | (pamt << 25)),                    // pad_amount
             (int)(dim0 << 16),                    // tensor_dim0[15:0]
             (int)((dim0 >> 16) | (rows << 16)),   // tensor_dim0[31:16], dim1[15:0]
             (int)((rows >> 16) | (dim0 << 16)),   // tensor_dim1[31:16], tile_dim0
             (int)rows,                            // tile_dim1 (tile_dim2 = 0)
             (int)stride,                          // tensor_dim0_stride[31:0]
             0, 0 };
  v4i gz = { 0, 0, 0, 0 };
#if __clang_major__ >= 23
  v8i gz8 = { 0, 0, 0, 0, 0, 0, 0, 0 };
  __builtin_amdgcn_tensor_load_to_lds(g0, g1, gz, gz, gz8, 0);
#else
  __builtin_amdgcn_tensor_load_to_lds(g0, g1, gz, gz, 0);
#endif
}
#endif

__device__ __forceinline__ unsigned ordkey(float a) {
  unsigned u = __float_as_uint(a);
  return (u & 0x80000000u) ? ~u : (u | 0x80000000u);
}

// ---------------- LayerNorm + |q|^2 ----------------
__global__ __launch_bounds__(256)
void ln_kernel(const float* __restrict__ q, const float* __restrict__ gamma,
               const float* __restrict__ beta, float* __restrict__ qln,
               float* __restrict__ q2g) {
  __shared__ float red[256];
  const int b = blockIdx.x, t = threadIdx.x;
  float x = q[(size_t)b * D_ + t];
  red[t] = x; __syncthreads();
  for (int k = 128; k > 0; k >>= 1) { if (t < k) red[t] += red[t + k]; __syncthreads(); }
  float mu = red[0] * (1.0f / D_); __syncthreads();
  float xc = x - mu;
  red[t] = xc * xc; __syncthreads();
  for (int k = 128; k > 0; k >>= 1) { if (t < k) red[t] += red[t + k]; __syncthreads(); }
  float var = red[0] * (1.0f / D_); __syncthreads();
  float y = xc * rsqrtf(var + 1e-5f) * gamma[t] + beta[t];
  qln[(size_t)b * D_ + t] = y;
  red[t] = y * y; __syncthreads();
  for (int k = 128; k > 0; k >>= 1) { if (t < k) red[t] += red[t + k]; __syncthreads(); }
  if (t == 0) q2g[b] = red[0];
}

// ---------------- keys: f32 -> f16 copy + |k|^2 ----------------
__global__ __launch_bounds__(256)
void cvtk_kernel(const float* __restrict__ keys, _Float16* __restrict__ k16,
                 float* __restrict__ k2g) {
  __shared__ float red[256];
  const int s = blockIdx.x, d = threadIdx.x;
  float v = keys[(size_t)s * D_ + d];
  k16[(size_t)s * D_ + d] = (_Float16)v;
  red[d] = v * v; __syncthreads();
  for (int k = 128; k > 0; k >>= 1) { if (d < k) red[d] += red[d + k]; __syncthreads(); }
  if (d == 0) k2g[s] = red[0];
}

// ---------------- values: f32 -> f16 tiled transpose (vT16[v][s]) ----------------
__global__ __launch_bounds__(256)
void cvtv_kernel(const float* __restrict__ values, _Float16* __restrict__ vT16) {
  __shared__ _Float16 t_s[64 * 72];
  const int s0 = blockIdx.x * 64, v0 = blockIdx.y * 64;
  const int t = threadIdx.x;
  {
    const int i = t & 63, qd = t >> 6;   // slot i, 16-dim group qd
    const float* src = values + (size_t)(s0 + i) * V_ + v0 + qd * 16;
#pragma unroll
    for (int c = 0; c < 4; ++c) {
      f4 v = ((const f4*)src)[c];
      const int vd = qd * 16 + c * 4;
      t_s[(vd + 0) * 72 + i] = (_Float16)v.x;
      t_s[(vd + 1) * 72 + i] = (_Float16)v.y;
      t_s[(vd + 2) * 72 + i] = (_Float16)v.z;
      t_s[(vd + 3) * 72 + i] = (_Float16)v.w;
    }
  }
  __syncthreads();
  {
    const int r = t >> 2, part = t & 3;  // vdim row r, 16-half segment part
    const _Float16* src = t_s + r * 72 + part * 16;
    _Float16* dst = vT16 + (size_t)(v0 + r) * S_ + s0 + part * 16;
    *(h8*)dst = *(const h8*)src;
    *(h8*)(dst + 8) = *(const h8*)(src + 8);
  }
}

// ---------------- Fused split-S flash attention (f16 WMMA + TDM K/V staging) ----------------
__global__ __launch_bounds__(256)
void attn_kernel(const float* __restrict__ qln, const float* __restrict__ q2g,
                 const _Float16* __restrict__ k16, const _Float16* __restrict__ vT16,
                 const float* __restrict__ k2g, const float* __restrict__ values,
                 float* __restrict__ part) {
  __shared__ __align__(16) char smem[SMEM_BYTES];
  _Float16* q_s    = (_Float16*)(smem + OFF_QS);
  _Float16* k_buf0 = (_Float16*)(smem + OFF_K0);
  _Float16* k_buf1 = (_Float16*)(smem + OFF_K1);
  _Float16* vT_s   = (_Float16*)(smem + OFF_VT);
  _Float16* p_s    = (_Float16*)(smem + OFF_PS);
  float*    mrg_o  = (float*)   (smem + OFF_MRGO);
  float*    mrg_ml = (float*)   (smem + OFF_MRGML);

  const int tid  = threadIdx.x;
  const int w    = tid >> 5;
  const int lane = tid & 31;
  const int h16  = lane >> 4;
  const int ln   = lane & 15;
  const int mblk = w & 3;          // which 16-row block of the 64-row tile
  const int nh   = w >> 2;         // which 32-slot half of the 64-slot strip
  const int nl0  = nh * 32;
  const int chunk = blockIdx.x, btile = blockIdx.y;
  const int row0  = btile * BT;

  // ---- stage q tile (64 x 256, f32 -> f16) ----
  {
    const int i = tid & 63, qd = tid >> 6;
    const float* src = qln + (size_t)(row0 + i) * D_ + qd * 64;
    _Float16* dst = q_s + i * QS_STRIDE + qd * 64;
#pragma unroll
    for (int jj = 0; jj < 16; ++jj) {
      f4 v = ((const f4*)src)[jj];
      dst[jj * 4 + 0] = (_Float16)v.x;
      dst[jj * 4 + 1] = (_Float16)v.y;
      dst[jj * 4 + 2] = (_Float16)v.z;
      dst[jj * 4 + 3] = (_Float16)v.w;
    }
  }
  float q2v[8];
#pragma unroll
  for (int j = 0; j < 8; ++j)
    q2v[j] = q2g[row0 + mblk * 16 + j + 8 * h16];

  v8f zero8 = {0.f, 0.f, 0.f, 0.f, 0.f, 0.f, 0.f, 0.f};
  v8f o[16];
#pragma unroll
  for (int t = 0; t < 16; ++t) o[t] = zero8;
  float m_run[8], l_run[8];
#pragma unroll
  for (int j = 0; j < 8; ++j) { m_run[j] = -3.0e38f; l_run[j] = 0.f; }

#ifdef HAVE_TDM
  const unsigned koff[2] = { (unsigned)(uintptr_t)(smem + OFF_K0),
                             (unsigned)(uintptr_t)(smem + OFF_K1) };
  const unsigned voff = (unsigned)(uintptr_t)(smem + OFF_VT);
  if (w == 0)   // prologue: DMA first K tile into buffer 0
    tdm_load_2d(k16 + (size_t)(chunk * SC) * D_, koff[0], 256u, 64u, 256u, 6u, 3u);
#endif

  for (int strip = 0; strip < NSTRIP; ++strip) {
    const int sbase = chunk * SC + strip * STRIP;
    __syncthreads();   // previous strip's compute done -> buffers reusable

#ifdef HAVE_TDM
    if (w == 0) {
      // V(strip): 256 rows x 64 halves, global row stride S, pad 12 DW / 32 DW
      tdm_load_2d(vT16 + sbase, voff, 64u, 256u, (unsigned)S_, 4u, 11u);
      if (strip + 1 < NSTRIP)   // K(strip+1): 64 rows x 256 halves, pad 4 DW / 128 DW
        tdm_load_2d(k16 + (size_t)(sbase + STRIP) * D_, koff[(strip + 1) & 1],
                    256u, 64u, 256u, 6u, 3u);
      // in-order TENSORcnt: leaving {V(s) [,K(s+1)]} in flight => K(s) resident
      if (strip + 1 < NSTRIP) __builtin_amdgcn_s_wait_tensorcnt(2);
      else                    __builtin_amdgcn_s_wait_tensorcnt(1);
    }
    if (strip + 1 < NSTRIP && tid < 64)   // emits global_prefetch_b8
      __builtin_prefetch(k2g + sbase + STRIP + tid, 0, 0);
#else
    {  // fallback: manual f16 copy of this strip's K tile
      _Float16* kd0 = ((strip & 1) ? k_buf1 : k_buf0);
      const int i = tid & 63, qd = tid >> 6;
      const _Float16* ksrc = k16 + (size_t)(sbase + i) * D_ + qd * 64;
      _Float16* kdst = kd0 + i * KS_STRIDE + qd * 64;
#pragma unroll
      for (int jj = 0; jj < 8; ++jj)
        *(h8*)(kdst + jj * 8) = *(const h8*)(ksrc + jj * 8);
    }
    {  // fallback: manual transpose staging of V from f32
      const int i = tid & 63, qd = tid >> 6;
      const float* vs = values + (size_t)(sbase + i) * V_ + qd * 64;
#pragma unroll
      for (int jj = 0; jj < 16; ++jj) {
        f4 v = ((const f4*)vs)[jj];
        const int vd = qd * 64 + jj * 4;
        vT_s[(vd + 0) * VT_STRIDE + i] = (_Float16)v.x;
        vT_s[(vd + 1) * VT_STRIDE + i] = (_Float16)v.y;
        vT_s[(vd + 2) * VT_STRIDE + i] = (_Float16)v.z;
        vT_s[(vd + 3) * VT_STRIDE + i] = (_Float16)v.w;
      }
    }
#endif
    __syncthreads();

    const _Float16* k_s = (strip & 1) ? k_buf1 : k_buf0;

    // ---- scores: 16(M) x 32(N) per wave via v_wmma_f32_16x16x32_f16 ----
    v8f c0 = zero8, c1 = zero8;
#pragma unroll
    for (int ks8 = 0; ks8 < 8; ++ks8) {
      const int k0 = ks8 * 32;
      const _Float16* qp = q_s + (mblk * 16 + ln) * QS_STRIDE + k0 + 16 * h16;
      v16h a; ((h8*)&a)[0] = *(const h8*)qp; ((h8*)&a)[1] = *(const h8*)(qp + 8);
      const _Float16* kp0 = k_s + (nl0 + ln) * KS_STRIDE + k0 + 16 * h16;
      v16h b0; ((h8*)&b0)[0] = *(const h8*)kp0; ((h8*)&b0)[1] = *(const h8*)(kp0 + 8);
      const _Float16* kp1 = k_s + (nl0 + 16 + ln) * KS_STRIDE + k0 + 16 * h16;
      v16h b1; ((h8*)&b1)[0] = *(const h8*)kp1; ((h8*)&b1)[1] = *(const h8*)(kp1 + 8);
      c0 = wmma16(a, b0, c0);
      c1 = wmma16(a, b1, c1);
    }
    const float k2a = k2g[sbase + nl0 + ln];
    const float k2b = k2g[sbase + nl0 + 16 + ln];

    // ---- online softmax (per 16-lane row groups) ----
    float scalef[8];
#pragma unroll
    for (int j = 0; j < 8; ++j) {
      float l0 = -fmaxf(q2v[j] + k2a - 2.0f * c0[j], 0.0f);
      float l1 = -fmaxf(q2v[j] + k2b - 2.0f * c1[j], 0.0f);
      float mx = fmaxf(l0, l1);
#pragma unroll
      for (int off = 8; off; off >>= 1) mx = fmaxf(mx, __shfl_xor(mx, off, 16));
      float mn = fmaxf(m_run[j], mx);
      float sc = __expf(m_run[j] - mn);
      float p0 = __expf(l0 - mn), p1 = __expf(l1 - mn);
      float rs = p0 + p1;
#pragma unroll
      for (int off = 8; off; off >>= 1) rs += __shfl_xor(rs, off, 16);
      l_run[j] = l_run[j] * sc + rs;
      m_run[j] = mn;
      scalef[j] = sc;
      _Float16* prow = p_s + (w * 16 + j + 8 * h16) * PS_STRIDE;
      prow[ln] = (_Float16)p0;
      prow[16 + ln] = (_Float16)p1;
    }
#pragma unroll
    for (int t = 0; t < 16; ++t)
#pragma unroll
      for (int j = 0; j < 8; ++j) o[t][j] *= scalef[j];

#ifdef HAVE_TDM
    if (w == 0) {   // V(strip) must be resident before P@V
      if (strip + 1 < NSTRIP) __builtin_amdgcn_s_wait_tensorcnt(1);
      else                    __builtin_amdgcn_s_wait_tensorcnt(0);
    }
    __syncthreads();
#endif

    // ---- P @ V : 16 output tiles of 16x16, K=32 slots ----
    const _Float16* pp = p_s + (w * 16 + ln) * PS_STRIDE + 16 * h16;
    v16h pa; ((h8*)&pa)[0] = *(const h8*)pp; ((h8*)&pa)[1] = *(const h8*)(pp + 8);
#pragma unroll
    for (int t = 0; t < 16; ++t) {
      const _Float16* vp = vT_s + (t * 16 + ln) * VT_STRIDE + nl0 + 16 * h16;
      v16h vb; ((h8*)&vb)[0] = *(const h8*)vp; ((h8*)&vb)[1] = *(const h8*)(vp + 8);
      o[t] = wmma16(pa, vb, o[t]);
    }
  }

  // ---- merge the two waves covering the same rows, write chunk partials ----
  __syncthreads();
  if (nh == 1) {
#pragma unroll
    for (int t = 0; t < 16; ++t)
#pragma unroll
      for (int j = 0; j < 8; ++j)
        mrg_o[(mblk * 16 + j + 8 * h16) * 256 + t * 16 + ln] = o[t][j];
    if (ln == 0) {
#pragma unroll
      for (int j = 0; j < 8; ++j) {
        const int r = mblk * 16 + j + 8 * h16;
        mrg_ml[r * 2 + 0] = m_run[j];
        mrg_ml[r * 2 + 1] = l_run[j];
      }
    }
  }
  __syncthreads();
  if (nh == 0) {
    float e1s[8], e2s[8];
#pragma unroll
    for (int j = 0; j < 8; ++j) {
      const int r = mblk * 16 + j + 8 * h16;
      float m2 = mrg_ml[r * 2 + 0], l2 = mrg_ml[r * 2 + 1];
      float mm = fmaxf(m_run[j], m2);
      float e1 = __expf(m_run[j] - mm), e2 = __expf(m2 - mm);
      e1s[j] = e1; e2s[j] = e2;
      if (ln == 0) {
        size_t base = ((size_t)chunk * B_ + row0 + r) * 258;
        part[base + 256] = mm;
        part[base + 257] = l_run[j] * e1 + l2 * e2;
      }
    }
#pragma unroll
    for (int t = 0; t < 16; ++t)
#pragma unroll
      for (int j = 0; j < 8; ++j) {
        const int r = mblk * 16 + j + 8 * h16;
        size_t base = ((size_t)chunk * B_ + row0 + r) * 258;
        part[base + t * 16 + ln] =
            o[t][j] * e1s[j] + mrg_o[r * 256 + t * 16 + ln] * e2s[j];
      }
  }
}

// ---------------- combine chunk partials -> retrieved + surprise ----------------
__global__ __launch_bounds__(256)
void combine_kernel(const float* __restrict__ part, const float* __restrict__ vt,
                    float* __restrict__ out_retr, float* __restrict__ out_surp) {
  __shared__ float red[256];
  const int b = blockIdx.x, v = threadIdx.x;
  float mm = -3.0e38f;
  for (int c = 0; c < NSC; ++c)
    mm = fmaxf(mm, part[((size_t)c * B_ + b) * 258 + 256]);
  float lt = 0.f, acc = 0.f;
  for (int c = 0; c < NSC; ++c) {
    size_t base = ((size_t)c * B_ + b) * 258;
    float e = __expf(part[base + 256] - mm);
    lt += part[base + 257] * e;
    acc += part[base + v] * e;
  }
  float r = acc / lt;
  out_retr[(size_t)b * V_ + v] = r;
  float d = r - vt[(size_t)b * V_ + v];
  red[v] = d * d; __syncthreads();
  for (int k = 128; k > 0; k >>= 1) { if (v < k) red[v] += red[v + k]; __syncthreads(); }
  if (v == 0) out_surp[b] = red[0] * (1.0f / V_);
}

__global__ __launch_bounds__(256)
void mean_kernel(const float* __restrict__ surp, float* __restrict__ meanp) {
  __shared__ float red[256];
  const int t = threadIdx.x;
  red[t] = surp[t] + surp[t + 256] + surp[t + 512] + surp[t + 768];
  __syncthreads();
  for (int k = 128; k > 0; k >>= 1) { if (t < k) red[t] += red[t + k]; __syncthreads(); }
  if (t == 0) meanp[0] = red[0] * (1.0f / B_);
}

// ---------------- top-k via ordered-float histogram + candidate ranking ----------------
__global__ void hzero_kernel(unsigned* hist, unsigned* ccount) {
  int i = blockIdx.x * 256 + threadIdx.x;
  hist[i] = 0u;
  if (i == 0) *ccount = 0u;
}

__global__ void hist_kernel(const float* __restrict__ age, unsigned* hist) {
  int s = blockIdx.x * 256 + threadIdx.x;
  atomicAdd(&hist[ordkey(age[s]) >> 16], 1u);
}

__global__ __launch_bounds__(256)
void scan_kernel(const unsigned* __restrict__ hist, unsigned* sel) {
  __shared__ unsigned seg[256];
  __shared__ unsigned above[256];
  const int t = threadIdx.x;
  unsigned s = 0;
  for (int i = 0; i < 256; ++i) s += hist[t * 256 + i];
  seg[t] = s; __syncthreads();
  if (t == 0) {
    unsigned a = 0;
    for (int i = 255; i >= 0; --i) { above[i] = a; a += seg[i]; }
  }
  __syncthreads();
  unsigned run = above[t];
  for (int i = 255; i >= 0; --i) {
    unsigned h = hist[t * 256 + i];
    if (run < NREP && run + h >= NREP) { sel[0] = (unsigned)(t * 256 + i); sel[1] = run; }
    run += h;
  }
}

__global__ void cand_kernel(const float* __restrict__ age, const unsigned* sel,
                            unsigned* ccount, unsigned* cand) {
  int s = blockIdx.x * 256 + threadIdx.x;
  if ((ordkey(age[s]) >> 16) >= sel[0]) {
    unsigned p = atomicAdd(ccount, 1u);
    cand[p] = (unsigned)s;
  }
}

__global__ __launch_bounds__(256)
void rank_kernel(const float* __restrict__ age, const unsigned* __restrict__ cand,
                 const unsigned* __restrict__ ccount, unsigned* __restrict__ oldest) {
  __shared__ unsigned red[256];
  const int t = threadIdx.x;
  const unsigned nc = *ccount;
  for (unsigned ci = blockIdx.x; ci < nc; ci += gridDim.x) {
    unsigned s = cand[ci];
    float as = age[s];
    unsigned cnt = 0;
    for (int sp = t; sp < S_; sp += 256) {
      float ap = age[sp];
      cnt += (unsigned)((ap > as) || (ap == as && (unsigned)sp < s));
    }
    red[t] = cnt; __syncthreads();
    for (int k = 128; k > 0; k >>= 1) { if (t < k) red[t] += red[t + k]; __syncthreads(); }
    if (t == 0 && red[0] < NREP) oldest[red[0]] = s;
    __syncthreads();
  }
}

// ---------------- base copies + gated scatter writes ----------------
__global__ void copy4_kernel(const float* __restrict__ src, float* __restrict__ dst, int n4) {
  int i = blockIdx.x * blockDim.x + threadIdx.x;
  for (; i < n4; i += gridDim.x * blockDim.x)
    ((f4*)dst)[i] = ((const f4*)src)[i];
}

__global__ void age_kernel(const float* __restrict__ age, float* __restrict__ out_age) {
  int s = blockIdx.x * 256 + threadIdx.x;
  out_age[s] = age[s] + 1.0f;
}

__global__ __launch_bounds__(256)
void apply_kernel(const float* __restrict__ keys, const float* __restrict__ values,
                  const float* __restrict__ qln, const float* __restrict__ vt,
                  const float* __restrict__ surp, const float* __restrict__ meanp,
                  const unsigned* __restrict__ oldest,
                  float* __restrict__ out_keys, float* __restrict__ out_vals,
                  float* __restrict__ out_age) {
  const int i = blockIdx.x;        // 0..NREP-1
  const int d = threadIdx.x;       // 0..255
  const unsigned s = oldest[i];
  float w = 1.0f / (1.0f + __expf(-(surp[i] - meanp[0])));
  float decay = 0.99f * (1.0f - w);
  float om = 1.0f - decay;
  out_keys[(size_t)s * D_ + d] = decay * keys[(size_t)s * D_ + d] + om * qln[(size_t)i * D_ + d];
  out_vals[(size_t)s * V_ + d] = decay * values[(size_t)s * V_ + d] + om * vt[(size_t)i * V_ + d];
  if (d == 0) out_age[s] = 1.0f;
}

extern "C" void kernel_launch(void* const* d_in, const int* in_sizes, int n_in,
                              void* d_out, int out_size, void* d_ws, size_t ws_size,
                              hipStream_t stream) {
  (void)in_sizes; (void)n_in; (void)out_size; (void)ws_size;
  const float* q      = (const float*)d_in[0];
  const float* vt     = (const float*)d_in[1];
  const float* keys   = (const float*)d_in[2];
  const float* values = (const float*)d_in[3];
  const float* age    = (const float*)d_in[4];
  const float* gamma  = (const float*)d_in[5];
  const float* beta   = (const float*)d_in[6];

  float* out = (float*)d_out;
  float* out_retr = out;                         // B*V
  float* out_surp = out + 262144;                // B
  float* out_keys = out + 263168;                // S*D
  float* out_vals = out + 17040384;              // S*V
  float* out_age  = out + 33817600;              // S

  char* ws = (char*)d_ws;
  size_t off = 0;
  float* qln  = (float*)(ws + off); off += (size_t)B_ * D_ * 4;        // 1 MB
  float* q2g  = (float*)(ws + off); off += (size_t)B_ * 4;             // 4 KB
  float* part = (float*)(ws + off); off += (size_t)NSC * B_ * 258 * 4; // 16.9 MB
  unsigned* hist   = (unsigned*)(ws + off); off += 65536 * 4;
  float*    meanp  = (float*)(ws + off);
  unsigned* ccount = (unsigned*)(ws + off + 64); off += 256;
  unsigned* sel    = (unsigned*)(ws + off); off += 256;
  unsigned* cand   = (unsigned*)(ws + off); off += 65536 * 4;
  unsigned* oldest = (unsigned*)(ws + off); off += NREP * 4;
  _Float16* k16    = (_Float16*)(ws + off); off += (size_t)S_ * D_ * 2; // 32 MB
  float*    k2g    = (float*)(ws + off); off += (size_t)S_ * 4;         // 256 KB
  _Float16* vT16   = (_Float16*)(ws + off); off += (size_t)S_ * V_ * 2; // 32 MB

  // read path prep
  ln_kernel<<<B_, 256, 0, stream>>>(q, gamma, beta, qln, q2g);
  cvtk_kernel<<<S_, 256, 0, stream>>>(keys, k16, k2g);
  cvtv_kernel<<<dim3(S_ / 64, V_ / 64), 256, 0, stream>>>(values, vT16);

  // top-k (independent of attention)
  hzero_kernel<<<256, 256, 0, stream>>>(hist, ccount);
  hist_kernel<<<S_ / 256, 256, 0, stream>>>(age, hist);
  scan_kernel<<<1, 256, 0, stream>>>(hist, sel);
  cand_kernel<<<S_ / 256, 256, 0, stream>>>(age, sel, ccount, cand);
  rank_kernel<<<4096, 256, 0, stream>>>(age, cand, ccount, oldest);

  // fused attention (WMMA + TDM) + combine
  attn_kernel<<<dim3(NSC, NBT), 256, 0, stream>>>(qln, q2g, k16, vT16, k2g, values, part);
  combine_kernel<<<B_, 256, 0, stream>>>(part, vt, out_retr, out_surp);
  mean_kernel<<<1, 256, 0, stream>>>(out_surp, meanp);

  // write path
  copy4_kernel<<<4096, 256, 0, stream>>>(keys, out_keys, (S_ * D_) / 4);
  copy4_kernel<<<4096, 256, 0, stream>>>(values, out_vals, (S_ * V_) / 4);
  age_kernel<<<S_ / 256, 256, 0, stream>>>(age, out_age);
  apply_kernel<<<NREP, 256, 0, stream>>>(keys, values, qln, vt, out_surp, meanp,
                                         oldest, out_keys, out_vals, out_age);
}
